// HybridEdgeNet_45526653337873
// MI455X (gfx1250) — compile-verified, matrix operand order checked by reading, not run on hardware
//
#include <hip/hip_runtime.h>

// ---------------------------------------------------------------------------
// HybridEdgeNet for MI455X (gfx1250, wave32, WMMA).
// Heavy path: two per-edge MLPs done as 16-row WMMA tiles
// (v_wmma_f32_16x16x32_f16, f32 accumulate), activations staged in LDS.
// ---------------------------------------------------------------------------

typedef __attribute__((ext_vector_type(16))) _Float16 v16h;
typedef __attribute__((ext_vector_type(8)))  _Float16 v8h;
typedef __attribute__((ext_vector_type(8)))  float    v8f;

#define DEVINL __device__ __forceinline__

// Problem constants (match reference)
constexpr int  BGRAPH = 512;
constexpr int  NPGc   = 128;
constexpr int  KDEG   = 8;
constexpr long NNODES = (long)BGRAPH * NPGc;   // 65536
constexpr long NEDGES = NNODES * KDEG;         // 524288

// ---------------------------------------------------------------------------
// WMMA MLP layer: 16 rows (edges) x KP -> NT*16, f16 in, f32 acc, f16 out.
// src: LDS tile [16][KP] (row-major, k contiguous, KP multiple of 32)
// W  : global f16 [NT*16][KP] (n-major, k contiguous)  (= W[n][k], i.e. x@W.T)
// dst: LDS tile [16][NT*16]
// ---------------------------------------------------------------------------
template <int KP, int NT, bool RELU>
DEVINL void mlp_layer(const _Float16* __restrict__ W,
                      const float* __restrict__ bias,
                      const _Float16* src, _Float16* dst, int lane)
{
    constexpr int KT = KP / 32;
    const int nl   = lane & 15;
    const int koff = (lane & 16) ? 8 : 0;   // f16 A/B lane layout: hi lanes hold k+8
    const int moff = (lane & 16) ? 8 : 0;   // f32 D layout: hi lanes hold m+8

    v16h a[KT];
#pragma unroll
    for (int t = 0; t < KT; ++t) {
        const _Float16* pr = src + nl * KP + t * 32 + koff;
        v8h lo = *(const v8h*)pr;
        v8h hi = *(const v8h*)(pr + 16);
        a[t] = __builtin_shufflevector(lo, hi, 0,1,2,3,4,5,6,7,
                                                8,9,10,11,12,13,14,15);
    }
#pragma unroll
    for (int nt = 0; nt < NT; ++nt) {
        v8f acc = {};
#pragma unroll
        for (int t = 0; t < KT; ++t) {
            const _Float16* wr = W + (size_t)(nt * 16 + nl) * KP + t * 32 + koff;
            v8h lo = *(const v8h*)wr;
            v8h hi = *(const v8h*)(wr + 16);
            v16h b = __builtin_shufflevector(lo, hi, 0,1,2,3,4,5,6,7,
                                                      8,9,10,11,12,13,14,15);
            acc = __builtin_amdgcn_wmma_f32_16x16x32_f16(
                false, a[t], false, b, (short)0, acc, false, false);
        }
        const float bn = bias[nt * 16 + nl];
#pragma unroll
        for (int r = 0; r < 8; ++r) {
            float v = acc[r] + bn;
            if (RELU) v = fmaxf(v, 0.0f);
            dst[(moff + r) * (NT * 16) + nt * 16 + nl] = (_Float16)v;
        }
    }
}

// ---------------------------------------------------------------------------
// BatchNorm statistics: sum / sumsq per feature (3 features) via atomics.
// ---------------------------------------------------------------------------
__global__ void bn_reduce_kernel(const float* __restrict__ x, float* __restrict__ acc, long N)
{
    float s[3] = {0.f, 0.f, 0.f}, ss[3] = {0.f, 0.f, 0.f};
    for (long i = (long)blockIdx.x * blockDim.x + threadIdx.x; i < N;
         i += (long)gridDim.x * blockDim.x) {
#pragma unroll
        for (int f = 0; f < 3; ++f) {
            float v = x[i * 3 + f];
            s[f] += v; ss[f] += v * v;
        }
    }
    __shared__ float red[256 * 6];
#pragma unroll
    for (int f = 0; f < 3; ++f) {
        red[threadIdx.x * 6 + f]     = s[f];
        red[threadIdx.x * 6 + 3 + f] = ss[f];
    }
    __syncthreads();
    for (int off = 128; off > 0; off >>= 1) {
        if ((int)threadIdx.x < off)
            for (int j = 0; j < 6; ++j)
                red[threadIdx.x * 6 + j] += red[(threadIdx.x + off) * 6 + j];
        __syncthreads();
    }
    if (threadIdx.x == 0)
        for (int j = 0; j < 6; ++j) atomicAdd(&acc[j], red[j]);
}

__global__ void bn_finalize_kernel(const float* __restrict__ acc,
                                   const float* __restrict__ g,
                                   const float* __restrict__ b,
                                   float* __restrict__ aff)
{
    int f = threadIdx.x;
    if (f < 3) {
        const float invN = 1.0f / (float)NNODES;
        float mean = acc[f] * invN;
        float var  = acc[3 + f] * invN - mean * mean;
        float sc   = g[f] * rsqrtf(var + 1e-5f);
        aff[f]     = sc;
        aff[3 + f] = b[f] - mean * sc;
    }
}

// f32 -> f16, zero-padded to [NP][KP] (n-major, k contiguous)
__global__ void convert_pad_kernel(_Float16* __restrict__ dst, const float* __restrict__ src,
                                   int Nr, int Kr, int NP, int KP)
{
    int i = blockIdx.x * blockDim.x + threadIdx.x;
    if (i >= NP * KP) return;
    int n = i / KP, k = i % KP;
    float v = (n < Nr && k < Kr) ? src[n * Kr + k] : 0.0f;
    dst[i] = (_Float16)v;
}

__global__ void pad_bias_kernel(float* __restrict__ dst, const float* __restrict__ src, int n)
{
    int t = threadIdx.x;
    if (t < 16) dst[t] = (t < n) ? src[t] : 0.0f;
}

// ---------------------------------------------------------------------------
// Encoder: per-edge MLP 6->64->128->64->16 (all ReLU) + segment-mean(8) +
// np linear -> zp[N][16]. One wave = 16 edges = 2 nodes; WG = 4 waves.
// ---------------------------------------------------------------------------
__global__ __launch_bounds__(128) void encoder_kernel(
    const float* __restrict__ x, const float* __restrict__ ea,
    const float* __restrict__ aff,
    const _Float16* __restrict__ w0, const float* __restrict__ b0,
    const _Float16* __restrict__ w1, const float* __restrict__ b1,
    const _Float16* __restrict__ w2, const float* __restrict__ b2,
    const _Float16* __restrict__ w3, const float* __restrict__ b3,
    const float* __restrict__ npW, const float* __restrict__ npb,
    float* __restrict__ zp)
{
    __shared__ _Float16 lds[4][2][2048];   // per-wave ping-pong [16][<=128]
    __shared__ float    zsh[4][2][16];
    const int wave = threadIdx.x >> 5;
    const int lane = threadIdx.x & 31;
    _Float16* A  = lds[wave][0];
    _Float16* Bb = lds[wave][1];
    const long e0 = (long)blockIdx.x * 64 + wave * 16;

    // zero input tile [16][32]
    {
        v8h z = {};
        v8h* p = (v8h*)A;
        p[lane * 2]     = z;
        p[lane * 2 + 1] = z;
    }
    __syncthreads();
    // fill: k0..2 = batchnormed x[dst], k3..5 = edge_attr
    {
        const int m = lane >> 1, half = lane & 1;
        const long e = e0 + m, d = e >> 3;
        if (half == 0) {
#pragma unroll
            for (int f = 0; f < 3; ++f)
                A[m * 32 + f] = (_Float16)(x[d * 3 + f] * aff[f] + aff[3 + f]);
        } else {
#pragma unroll
            for (int f = 0; f < 3; ++f)
                A[m * 32 + 3 + f] = (_Float16)ea[e * 3 + f];
        }
    }
    __syncthreads();
    mlp_layer<32, 4, true>(w0, b0, A,  Bb, lane);  //   6(→32) ->  64
    __syncthreads();
    mlp_layer<64, 8, true>(w1, b1, Bb, A,  lane);  //  64      -> 128
    __syncthreads();
    mlp_layer<128, 4, true>(w2, b2, A,  Bb, lane); // 128      ->  64
    __syncthreads();
    mlp_layer<64, 1, true>(w3, b3, Bb, A,  lane);  //  64      ->  16 (final relu)
    __syncthreads();
    // segment mean over the 8 edges of each of the wave's 2 nodes
    {
        const int nl = lane >> 4, d = lane & 15;
        float s = 0.f;
#pragma unroll
        for (int r = 0; r < 8; ++r) s += (float)A[(nl * 8 + r) * 16 + d];
        zsh[wave][nl][d] = s * 0.125f;
    }
    __syncthreads();
    // zp = z @ np_W.T + np_b
    {
        const int nl = lane >> 4, d = lane & 15;
        float s = npb[d];
#pragma unroll
        for (int l2 = 0; l2 < 16; ++l2) s += zsh[wave][nl][l2] * npW[d * 16 + l2];
        const long node = (e0 >> 3) + nl;
        zp[node * 16 + d] = s;
    }
}

// ---------------------------------------------------------------------------
// Per-graph transformer (2 layers, 2 heads, DM=16, seq=129) + cross-attn.
// One 128-thread WG per graph; tiny FLOPs -> scalar VALU. Writes refined[B][16].
// ---------------------------------------------------------------------------
struct TfP {
    const float *x, *aff, *zp;
    float* refined;
    const float *embW, *embb, *cls;
    const float *inw[2], *inb[2], *outw[2], *outb[2];
    const float *f1w[2], *f1b[2], *f2w[2], *f2b[2];
    const float *ln1g[2], *ln1b[2], *ln2g[2], *ln2b[2];
    const float *cainw, *cainb, *caoutw, *caoutb, *c2nW, *c2nb;
};

__global__ __launch_bounds__(128) void transformer_kernel(TfP P)
{
    __shared__ float h[129 * 16];
    __shared__ float qkv[129 * 48];
    __shared__ float red[132];
    __shared__ float small[48];
    const int t = threadIdx.x;
    const int g = blockIdx.x;

    // embed nodes (+cls at row 0)
    {
        const long node = (long)g * NPGc + t;
        float xb[3];
#pragma unroll
        for (int f = 0; f < 3; ++f)
            xb[f] = P.x[node * 3 + f] * P.aff[f] + P.aff[3 + f];
#pragma unroll
        for (int d = 0; d < 16; ++d)
            h[(1 + t) * 16 + d] = P.embb[d] + P.embW[d * 3 + 0] * xb[0]
                                           + P.embW[d * 3 + 1] * xb[1]
                                           + P.embW[d * 3 + 2] * xb[2];
        if (t == 0)
            for (int d = 0; d < 16; ++d) h[d] = P.cls[d];
    }
    __syncthreads();

    for (int l = 0; l < 2; ++l) {
        // QKV projections
        for (int r = t; r < 129; r += 128) {
            float hr[16];
#pragma unroll
            for (int d = 0; d < 16; ++d) hr[d] = h[r * 16 + d];
            for (int j = 0; j < 48; ++j) {
                float s = P.inb[l][j];
#pragma unroll
                for (int d = 0; d < 16; ++d) s += P.inw[l][j * 16 + d] * hr[d];
                qkv[r * 48 + j] = s;
            }
        }
        __syncthreads();
        // attention (online softmax), out-proj, residual, LN1
        for (int r = t; r < 129; r += 128) {
            float o[16];
            for (int hh = 0; hh < 2; ++hh) {
                float q[8];
#pragma unroll
                for (int d = 0; d < 8; ++d) q[d] = qkv[r * 48 + hh * 8 + d];
                float mx = -3.0e38f, sum = 0.f, av[8];
#pragma unroll
                for (int d = 0; d < 8; ++d) av[d] = 0.f;
                for (int kk = 0; kk < 129; ++kk) {
                    float s = 0.f;
#pragma unroll
                    for (int d = 0; d < 8; ++d)
                        s += q[d] * qkv[kk * 48 + 16 + hh * 8 + d];
                    s *= 0.35355339059327373f;  // 1/sqrt(8)
                    float nm = fmaxf(mx, s);
                    float c = __expf(mx - nm), e = __expf(s - nm);
                    sum = sum * c + e;
#pragma unroll
                    for (int d = 0; d < 8; ++d)
                        av[d] = av[d] * c + e * qkv[kk * 48 + 32 + hh * 8 + d];
                    mx = nm;
                }
                float inv = 1.0f / sum;
#pragma unroll
                for (int d = 0; d < 8; ++d) o[hh * 8 + d] = av[d] * inv;
            }
            float hn[16];
#pragma unroll
            for (int d = 0; d < 16; ++d) {
                float s = P.outb[l][d];
#pragma unroll
                for (int e2 = 0; e2 < 16; ++e2) s += P.outw[l][d * 16 + e2] * o[e2];
                hn[d] = h[r * 16 + d] + s;
            }
            float m = 0.f;
#pragma unroll
            for (int d = 0; d < 16; ++d) m += hn[d];
            m *= 0.0625f;
            float v = 0.f;
#pragma unroll
            for (int d = 0; d < 16; ++d) { float dd = hn[d] - m; v += dd * dd; }
            v *= 0.0625f;
            float inv = rsqrtf(v + 1e-5f);
#pragma unroll
            for (int d = 0; d < 16; ++d)
                h[r * 16 + d] = (hn[d] - m) * inv * P.ln1g[l][d] + P.ln1b[l][d];
        }
        __syncthreads();
        // FFN + residual + LN2
        for (int r = t; r < 129; r += 128) {
            float hr[16];
#pragma unroll
            for (int d = 0; d < 16; ++d) hr[d] = h[r * 16 + d];
            float ff[32];
            for (int j = 0; j < 32; ++j) {
                float s = P.f1b[l][j];
#pragma unroll
                for (int d = 0; d < 16; ++d) s += P.f1w[l][j * 16 + d] * hr[d];
                ff[j] = fmaxf(s, 0.f);
            }
            float hn[16];
#pragma unroll
            for (int d = 0; d < 16; ++d) {
                float s = P.f2b[l][d];
#pragma unroll
                for (int j = 0; j < 32; ++j) s += P.f2w[l][d * 32 + j] * ff[j];
                hn[d] = hr[d] + s;
            }
            float m = 0.f;
#pragma unroll
            for (int d = 0; d < 16; ++d) m += hn[d];
            m *= 0.0625f;
            float v = 0.f;
#pragma unroll
            for (int d = 0; d < 16; ++d) { float dd = hn[d] - m; v += dd * dd; }
            v *= 0.0625f;
            float inv = rsqrtf(v + 1e-5f);
#pragma unroll
            for (int d = 0; d < 16; ++d)
                h[r * 16 + d] = (hn[d] - m) * inv * P.ln2g[l][d] + P.ln2b[l][d];
        }
        __syncthreads();
    }

    // cross attention: q from cls token, keys/values from zb = zp rows
    if (t < 16) {
        float s = P.cainb[t];
#pragma unroll
        for (int d = 0; d < 16; ++d) s += P.cainw[t * 16 + d] * h[d];
        small[t] = s;
    }
    __syncthreads();
    {
        const long node = (long)g * NPGc + t;
        float zb[16];
#pragma unroll
        for (int d = 0; d < 16; ++d) zb[d] = P.zp[node * 16 + d];
        float s = 0.f;
        for (int d = 0; d < 16; ++d) {
            float kj = P.cainb[16 + d];
#pragma unroll
            for (int e2 = 0; e2 < 16; ++e2) kj += P.cainw[(16 + d) * 16 + e2] * zb[e2];
            s += small[d] * kj;
        }
        red[t] = s * 0.25f;  // 1/sqrt(16)
        for (int d = 0; d < 16; ++d) {
            float vj = P.cainb[32 + d];
#pragma unroll
            for (int e2 = 0; e2 < 16; ++e2) vj += P.cainw[(32 + d) * 16 + e2] * zb[e2];
            qkv[t * 16 + d] = vj;
        }
    }
    __syncthreads();
    if (t == 0) {
        float m = -3.0e38f;
        for (int j = 0; j < 128; ++j) m = fmaxf(m, red[j]);
        red[128] = m;
    }
    __syncthreads();
    red[t] = __expf(red[t] - red[128]);
    __syncthreads();
    if (t == 0) {
        float s = 0.f;
        for (int j = 0; j < 128; ++j) s += red[j];
        red[129] = 1.0f / s;
    }
    __syncthreads();
    if (t < 16) {
        float s = 0.f;
        for (int j = 0; j < 128; ++j) s += red[j] * qkv[j * 16 + t];
        small[16 + t] = s * red[129];   // att
    }
    __syncthreads();
    if (t < 16) {
        float s = P.caoutb[t];
#pragma unroll
        for (int d = 0; d < 16; ++d) s += P.caoutw[t * 16 + d] * small[16 + d];
        small[32 + t] = s;              // out proj
    }
    __syncthreads();
    if (t < 16) {
        float s = P.c2nb[t];
#pragma unroll
        for (int d = 0; d < 16; ++d) s += P.c2nW[t * 16 + d] * small[32 + d];
        P.refined[g * 16 + t] = s;
    }
}

// ---------------------------------------------------------------------------
// Decoder: per-edge MLP 32->64->128->64->3 (no final relu) + segment-mean(8).
// Input tile: k0..15 = zp[dst]+refined[g], k16..31 = zp[src]-zp[dst].
// ---------------------------------------------------------------------------
__global__ __launch_bounds__(128) void decoder_kernel(
    const float* __restrict__ zp, const float* __restrict__ refined,
    const int* __restrict__ esrc,
    const _Float16* __restrict__ w0, const float* __restrict__ b0,
    const _Float16* __restrict__ w1, const float* __restrict__ b1,
    const _Float16* __restrict__ w2, const float* __restrict__ b2,
    const _Float16* __restrict__ w3, const float* __restrict__ b3,
    float* __restrict__ out)
{
    __shared__ _Float16 lds[4][2][2048];
    const int wave = threadIdx.x >> 5;
    const int lane = threadIdx.x & 31;
    _Float16* A  = lds[wave][0];
    _Float16* Bb = lds[wave][1];
    const long e0 = (long)blockIdx.x * 64 + wave * 16;

    {
        const int m = lane >> 1, half = lane & 1, kb = half * 8;
        const long e = e0 + m, dn = e >> 3;
        const long sn = (long)esrc[e];
        const long gg = dn >> 7;  // NPG = 128
#pragma unroll
        for (int j = 0; j < 8; ++j) {
            float zd = zp[dn * 16 + kb + j];
            float zs = zp[sn * 16 + kb + j];
            float rf = refined[gg * 16 + kb + j];
            A[m * 32 + kb + j]      = (_Float16)(zd + rf);
            A[m * 32 + 16 + kb + j] = (_Float16)(zs - zd);
        }
    }
    __syncthreads();
    mlp_layer<32, 4, true>(w0, b0, A,  Bb, lane);   //  32 ->  64
    __syncthreads();
    mlp_layer<64, 8, true>(w1, b1, Bb, A,  lane);   //  64 -> 128
    __syncthreads();
    mlp_layer<128, 4, true>(w2, b2, A,  Bb, lane);  // 128 ->  64
    __syncthreads();
    mlp_layer<64, 1, false>(w3, b3, Bb, A,  lane);  //  64 ->  16 (3 real, no relu)
    __syncthreads();
    {
        const int nl = lane >> 4, d = lane & 15;
        if (d < 3) {
            float s = 0.f;
#pragma unroll
            for (int r = 0; r < 8; ++r) s += (float)A[(nl * 8 + r) * 16 + d];
            const long node = (e0 >> 3) + nl;
            out[node * 3 + d] = s * 0.125f;
        }
    }
}

// ---------------------------------------------------------------------------
// Launch
// ---------------------------------------------------------------------------
extern "C" void kernel_launch(void* const* d_in, const int* in_sizes, int n_in,
                              void* d_out, int out_size, void* d_ws, size_t ws_size,
                              hipStream_t stream)
{
    (void)in_sizes; (void)n_in; (void)out_size; (void)ws_size;

    const float* x   = (const float*)d_in[0];
    const float* ea  = (const float*)d_in[1];
    const float* bng = (const float*)d_in[2];
    const float* bnb = (const float*)d_in[3];
    const float* encW[4] = {(const float*)d_in[4], (const float*)d_in[6],
                            (const float*)d_in[8], (const float*)d_in[10]};
    const float* encB[4] = {(const float*)d_in[5], (const float*)d_in[7],
                            (const float*)d_in[9], (const float*)d_in[11]};
    const float* npW  = (const float*)d_in[12];
    const float* npb  = (const float*)d_in[13];
    const float* decW[4] = {(const float*)d_in[47], (const float*)d_in[49],
                            (const float*)d_in[51], (const float*)d_in[53]};
    const float* decB[4] = {(const float*)d_in[48], (const float*)d_in[50],
                            (const float*)d_in[52], (const float*)d_in[54]};
    const int* eidx = (const int*)d_in[55];  // row 0 = src

    // workspace layout
    char* w = (char*)d_ws;
    float* acc = (float*)(w + 0);    // 6 floats (sum, sumsq)
    float* aff = (float*)(w + 64);   // 6 floats (scale, shift)
    float* db3 = (float*)(w + 128);  // 16 floats (padded dec_b3)
    size_t o = 256;
    auto take = [&](size_t bytes) { void* p = w + o; o += (bytes + 255) & ~(size_t)255; return p; };
    _Float16* wE0 = (_Float16*)take(64  * 32  * 2);
    _Float16* wE1 = (_Float16*)take(128 * 64  * 2);
    _Float16* wE2 = (_Float16*)take(64  * 128 * 2);
    _Float16* wE3 = (_Float16*)take(16  * 64  * 2);
    _Float16* wD0 = (_Float16*)take(64  * 32  * 2);
    _Float16* wD1 = (_Float16*)take(128 * 64  * 2);
    _Float16* wD2 = (_Float16*)take(64  * 128 * 2);
    _Float16* wD3 = (_Float16*)take(16  * 64  * 2);
    float* zp      = (float*)take((size_t)NNODES * 16 * 4);
    float* refined = (float*)take((size_t)BGRAPH * 16 * 4);

    hipMemsetAsync(d_ws, 0, 256, stream);

    bn_reduce_kernel<<<128, 256, 0, stream>>>(x, acc, NNODES);
    bn_finalize_kernel<<<1, 32, 0, stream>>>(acc, bng, bnb, aff);
    pad_bias_kernel<<<1, 16, 0, stream>>>(db3, decB[3], 3);

    convert_pad_kernel<<<8,  256, 0, stream>>>(wE0, encW[0], 64, 6,   64, 32);
    convert_pad_kernel<<<32, 256, 0, stream>>>(wE1, encW[1], 128, 64, 128, 64);
    convert_pad_kernel<<<32, 256, 0, stream>>>(wE2, encW[2], 64, 128, 64, 128);
    convert_pad_kernel<<<4,  256, 0, stream>>>(wE3, encW[3], 16, 64,  16, 64);
    convert_pad_kernel<<<8,  256, 0, stream>>>(wD0, decW[0], 64, 32,  64, 32);
    convert_pad_kernel<<<32, 256, 0, stream>>>(wD1, decW[1], 128, 64, 128, 64);
    convert_pad_kernel<<<32, 256, 0, stream>>>(wD2, decW[2], 64, 128, 64, 128);
    convert_pad_kernel<<<4,  256, 0, stream>>>(wD3, decW[3], 3, 64,   16, 64);

    encoder_kernel<<<(int)(NEDGES / 64), 128, 0, stream>>>(
        x, ea, aff, wE0, encB[0], wE1, encB[1], wE2, encB[2], wE3, encB[3],
        npW, npb, zp);

    TfP P;
    P.x = x; P.aff = aff; P.zp = zp; P.refined = refined;
    P.embW = (const float*)d_in[14];
    P.embb = (const float*)d_in[15];
    P.cls  = (const float*)d_in[16];
    for (int l = 0; l < 2; ++l) {
        int base = 17 + l * 12;
        P.inw[l]  = (const float*)d_in[base + 0];
        P.inb[l]  = (const float*)d_in[base + 1];
        P.outw[l] = (const float*)d_in[base + 2];
        P.outb[l] = (const float*)d_in[base + 3];
        P.f1w[l]  = (const float*)d_in[base + 4];
        P.f1b[l]  = (const float*)d_in[base + 5];
        P.f2w[l]  = (const float*)d_in[base + 6];
        P.f2b[l]  = (const float*)d_in[base + 7];
        P.ln1g[l] = (const float*)d_in[base + 8];
        P.ln1b[l] = (const float*)d_in[base + 9];
        P.ln2g[l] = (const float*)d_in[base + 10];
        P.ln2b[l] = (const float*)d_in[base + 11];
    }
    P.cainw  = (const float*)d_in[41];
    P.cainb  = (const float*)d_in[42];
    P.caoutw = (const float*)d_in[43];
    P.caoutb = (const float*)d_in[44];
    P.c2nW   = (const float*)d_in[45];
    P.c2nb   = (const float*)d_in[46];
    transformer_kernel<<<BGRAPH, 128, 0, stream>>>(P);

    decoder_kernel<<<(int)(NEDGES / 64), 128, 0, stream>>>(
        zp, refined, eidx, wD0, decB[0], wD1, decB[1], wD2, decB[2], wD3, db3,
        (float*)d_out);
}
